// Evo2Attention_24446953849297
// MI455X (gfx1250) — matrix operand order repaired and checked by reading, not compiled, not run on hardware
//
#include <hip/hip_runtime.h>
#include <hip/hip_bf16.h>
#include <math.h>
#include <stdint.h>

// ---------------- problem constants (from reference) ----------------
constexpr int B  = 2;
constexpr int S  = 2048;
constexpr int H  = 2048;
constexpr int NH = 16;
constexpr int HD = 128;
constexpr int M_ROWS = B * S;          // 4096
constexpr float INV_SQRT_HD = 0.08838834764831845f; // 1/sqrt(128)
constexpr float LN_THETA_OVER_HALF = 0.14391156832817f; // ln(10000)/64

typedef __bf16 bf16;
typedef __attribute__((ext_vector_type(16))) __bf16 v16bf;
typedef __attribute__((ext_vector_type(8)))  __bf16 v8bf;
typedef __attribute__((ext_vector_type(8)))  float  v8f;

static __device__ inline v16bf mk16(v8bf lo, v8bf hi) {
    v16bf r;
#pragma unroll
    for (int i = 0; i < 8; ++i) { r[i] = lo[i]; r[i + 8] = hi[i]; }
    return r;
}
static __device__ inline v8f zero8() {
    v8f z;
#pragma unroll
    for (int i = 0; i < 8; ++i) z[i] = 0.0f;
    return z;
}
static __device__ inline v8f wmma_bf16(v16bf a, v16bf b, v8f c) {
    // D = A(16x32) * B(32x16) + C, fp32 accum
    return __builtin_amdgcn_wmma_f32_16x16x32_bf16(false, a, false, b, (short)0, c, false, false);
}

// -------- CDNA5 async global->LDS copy (ASYNCcnt-tracked, VGPR-bypassing) --------
typedef __attribute__((address_space(3))) bf16 lds_bf16;
static __device__ inline uint32_t lds_off(const bf16* p) {
    return (uint32_t)(uintptr_t)(lds_bf16*)p;   // generic -> LDS addrspace -> 32-bit offset
}
// copy 16 bytes per lane
static __device__ inline void async_cp16(uint32_t lds, uint64_t gaddr) {
    asm volatile("global_load_async_to_lds_b128 %0, %1, off"
                 :: "v"(lds), "v"(gaddr) : "memory");
}
// copy 32 bytes per lane (two b128, INST_OFFSET applies to both LDS and global side)
static __device__ inline void async_cp32(uint32_t lds, uint64_t gaddr) {
    asm volatile("global_load_async_to_lds_b128 %0, %1, off\n\t"
                 "global_load_async_to_lds_b128 %0, %1, off offset:16"
                 :: "v"(lds), "v"(gaddr) : "memory");
}
// copy 64 bytes per lane (four b128)
static __device__ inline void async_cp64(uint32_t lds, uint64_t gaddr) {
    asm volatile("global_load_async_to_lds_b128 %0, %1, off\n\t"
                 "global_load_async_to_lds_b128 %0, %1, off offset:16\n\t"
                 "global_load_async_to_lds_b128 %0, %1, off offset:32\n\t"
                 "global_load_async_to_lds_b128 %0, %1, off offset:48"
                 :: "v"(lds), "v"(gaddr) : "memory");
}
static __device__ inline void wait_async0() {
    asm volatile("s_wait_asynccnt 0x0" ::: "memory");
}

// ---------------- fp32 -> bf16 convert ----------------
__global__ void cvt_f32_bf16(const float* __restrict__ in, bf16* __restrict__ out, int n) {
    int i = blockIdx.x * blockDim.x + threadIdx.x;
    if (i < n) out[i] = (bf16)in[i];
}

// ---------------- GEMM: C[M,N] = A[M,K] * W[N,K]^T  (bf16 in, f32 acc) ----------------
// 128x128 tile, BK=64 stage, double-buffered async global->LDS, 256 threads = 8 waves
// (4 along M x 2 along N), each wave: 2 M-subtiles x 4 N-subtiles of 16x16, 16 WMMA/stage.
template <bool OUT_F32>
__global__ __launch_bounds__(256) void gemm_xwt(const bf16* __restrict__ A,
                                                const bf16* __restrict__ W,
                                                void* __restrict__ Cout,
                                                int M, int N, int K) {
    __shared__ __align__(16) bf16 As[2][128][64];
    __shared__ __align__(16) bf16 Bs[2][128][64];   // Bs[.][n][k] = W[n0+n][k0+k]

    const int nBlocksN = N / 128;
    const int bm = (int)blockIdx.x / nBlocksN;
    const int bn = (int)blockIdx.x % nBlocksN;

    const int tid  = threadIdx.x;
    const int wave = tid >> 5;
    const int lane = tid & 31;
    const int lm   = lane & 15;
    const int h    = lane >> 4;
    const int wm   = wave & 3;   // 0..3  -> 32 rows each
    const int wn   = wave >> 2;  // 0..1  -> 64 cols each

    v8f acc[2][4];
#pragma unroll
    for (int i = 0; i < 2; ++i)
#pragma unroll
        for (int j = 0; j < 4; ++j) acc[i][j] = zero8();

    // async staging assignment: 2 threads per row, 32 elements (64B) each
    const int aRow = tid >> 1;           // 0..127
    const int aCol = (tid & 1) * 32;     // 0 or 32
    const bf16* agBase = A + (size_t)(bm * 128 + aRow) * K + aCol;
    const bf16* wgBase = W + (size_t)(bn * 128 + aRow) * K + aCol;

    const int NS = K / 64;
    // prologue: stage 0 in flight
    async_cp64(lds_off(&As[0][aRow][aCol]), (uint64_t)agBase);
    async_cp64(lds_off(&Bs[0][aRow][aCol]), (uint64_t)wgBase);

    for (int ks = 0; ks < NS; ++ks) {
        const int buf = ks & 1;
        wait_async0();          // stage ks landed in LDS (only stage in flight)
        __syncthreads();        // visible to all waves; prior reads of buf^1 done
        if (ks + 1 < NS) {      // overlap: issue stage ks+1 while computing ks
            async_cp64(lds_off(&As[buf ^ 1][aRow][aCol]), (uint64_t)(agBase + (ks + 1) * 64));
            async_cp64(lds_off(&Bs[buf ^ 1][aRow][aCol]), (uint64_t)(wgBase + (ks + 1) * 64));
        }
#pragma unroll
        for (int kk = 0; kk < 64; kk += 32) {
            // A-frags: lane holds row M=lm, elements i<8 -> k=i+8h, i>=8 -> k=16+(i-8)+8h
            v16bf af[2];
#pragma unroll
            for (int im = 0; im < 2; ++im) {
                const int m = wm * 32 + im * 16 + lm;
                af[im] = mk16(*(const v8bf*)&As[buf][m][kk + 8 * h],
                              *(const v8bf*)&As[buf][m][kk + 16 + 8 * h]);
            }
            // B-frags: lane holds col N=lm, elements i<8 -> k=i+8h, i>=8 -> k=16+(i-8)+8h
#pragma unroll
            for (int jn = 0; jn < 4; ++jn) {
                const int n = wn * 64 + jn * 16 + lm;
                v16bf bfrag = mk16(*(const v8bf*)&Bs[buf][n][kk + 8 * h],
                                   *(const v8bf*)&Bs[buf][n][kk + 16 + 8 * h]);
#pragma unroll
                for (int im = 0; im < 2; ++im)
                    acc[im][jn] = wmma_bf16(af[im], bfrag, acc[im][jn]);
            }
        }
    }

    // C layout: VGPR r -> row M = r + 8h ; lane -> col N = lm
    float* Cf = (float*)Cout;
    bf16*  Cb = (bf16*)Cout;
#pragma unroll
    for (int im = 0; im < 2; ++im)
#pragma unroll
        for (int jn = 0; jn < 4; ++jn)
#pragma unroll
            for (int r = 0; r < 8; ++r) {
                const int row = bm * 128 + wm * 32 + im * 16 + r + 8 * h;
                const int col = bn * 128 + wn * 64 + jn * 16 + lm;
                const size_t idx = (size_t)row * N + col;
                if constexpr (OUT_F32) Cf[idx] = acc[im][jn][r];
                else                   Cb[idx] = (bf16)acc[im][jn][r];
            }
}

// ---------------- RoPE + [B,S,NH,HD] -> [B*NH,S,HD] transpose ----------------
__global__ __launch_bounds__(128) void rope_transpose(const bf16* __restrict__ Qp,
                                                      const bf16* __restrict__ Kp,
                                                      const bf16* __restrict__ Vp,
                                                      bf16* __restrict__ Qr,
                                                      bf16* __restrict__ Kr,
                                                      bf16* __restrict__ Vr) {
    const int idx = blockIdx.x;
    const int s  = idx % S;
    const int hh = (idx / S) % NH;
    const int b  = idx / (S * NH);
    const int d  = threadIdx.x;               // 0..127
    const int f  = d & 63;
    const int dp = (d + 64) & 127;
    const float sign = (d < 64) ? -1.0f : 1.0f;

    const float inv_freq = __expf(-(float)f * LN_THETA_OVER_HALF);
    const float ang = (float)s * inv_freq;
    float c, si;
    __sincosf(ang, &si, &c);   // si=sin, c=cos

    const size_t src = (size_t)(b * S + s) * H + hh * HD;
    const size_t dst = ((size_t)(b * NH + hh) * S + s) * HD + d;

    const float qv = (float)Qp[src + d];
    const float qo = (float)Qp[src + dp];
    Qr[dst] = (bf16)(qv * c + sign * qo * si);

    const float kv = (float)Kp[src + d];
    const float ko = (float)Kp[src + dp];
    Kr[dst] = (bf16)(kv * c + sign * ko * si);

    Vr[dst] = Vp[src + d];
}

// ---------------- Flash attention (causal), bf16 WMMA, online softmax ----------------
// block = 256 threads = 8 waves; q-tile = 128 rows (16 per wave); key chunk = 32.
// K tile staged with async global->LDS; V staged transposed via VALU (async can't transpose).
__global__ __launch_bounds__(256) void flash_attn(const bf16* __restrict__ Q,
                                                  const bf16* __restrict__ K,
                                                  const bf16* __restrict__ V,
                                                  bf16* __restrict__ ctx) {
    __shared__ __align__(16) bf16 Ks[32][128];     // [key][d]
    __shared__ __align__(16) bf16 Vst[128][32];    // [d][key]  (transposed)
    __shared__ __align__(16) bf16 Ps[8][16][32];   // per-wave P tile [m][key]

    constexpr int QT = S / 128;                    // 16 q-tiles per (b,h)
    const int qt   = (int)blockIdx.x % QT;
    const int bh   = (int)blockIdx.x / QT;
    const int b    = bh / NH;
    const int head = bh % NH;

    const int tid  = threadIdx.x;
    const int wave = tid >> 5;
    const int lane = tid & 31;
    const int lm   = lane & 15;
    const int h    = lane >> 4;

    const bf16* Qp = Q + (size_t)bh * S * HD;
    const bf16* Kp = K + (size_t)bh * S * HD;
    const bf16* Vp = V + (size_t)bh * S * HD;

    // Load Q fragments: 16 rows x 128 d per wave, 4 chunks of K-dim 32
    const int qrow = qt * 128 + wave * 16 + lm;
    v16bf qf[4];
#pragma unroll
    for (int dk = 0; dk < 4; ++dk) {
        const bf16* qg = Qp + (size_t)qrow * HD + dk * 32;
        qf[dk] = mk16(*(const v8bf*)(qg + 8 * h),
                      *(const v8bf*)(qg + 16 + 8 * h));
    }

    v8f acc[8];
#pragma unroll
    for (int n = 0; n < 8; ++n) acc[n] = zero8();
    float m_run[8], l_run[8];
#pragma unroll
    for (int r = 0; r < 8; ++r) { m_run[r] = -1e30f; l_run[r] = 0.0f; }

    const int ldKey = tid >> 3;          // 0..31
    const int ldCol = (tid & 7) * 16;    // 0..112
    const uint32_t ksLds = lds_off(&Ks[ldKey][ldCol]);

    const int nkc = (qt + 1) * 4;        // 32-key chunks, causal bound
    for (int kc = 0; kc < nkc; ++kc) {
        const int key0 = kc * 32;
        __syncthreads();                 // previous chunk fully consumed
        // stage K [32][128] asynchronously; V [128][32] transposed via VALU
        async_cp32(ksLds, (uint64_t)(Kp + (size_t)(key0 + ldKey) * HD + ldCol));
        {
            const bf16* vg = Vp + (size_t)(key0 + ldKey) * HD + ldCol;
#pragma unroll
            for (int j = 0; j < 2; ++j) {
                v8bf vv = *(const v8bf*)(vg + 8 * j);
#pragma unroll
                for (int e = 0; e < 8; ++e) Vst[ldCol + 8 * j + e][ldKey] = vv[e];
            }
        }
        wait_async0();
        __syncthreads();

        // scores: S(16x32) = Q(16x128) * K^T(128x32), two 16-key subtiles
        v8f sc[2];
#pragma unroll
        for (int j = 0; j < 2; ++j) {
            sc[j] = zero8();
#pragma unroll
            for (int dk = 0; dk < 4; ++dk) {
                // B element (i,lane): kdim = i<8 ? i+8h : 16+(i-8)+8h (+32dk), col(key) = lm + 16j
                v16bf bf = mk16(*(const v8bf*)&Ks[16 * j + lm][32 * dk + 8 * h],
                                *(const v8bf*)&Ks[16 * j + lm][32 * dk + 16 + 8 * h]);
                sc[j] = wmma_bf16(qf[dk], bf, sc[j]);
            }
        }

        // scale + causal mask + online softmax; row M = r + 8h lives in lanes (16h..16h+15)
#pragma unroll
        for (int r = 0; r < 8; ++r) {
            const int qglob = qt * 128 + wave * 16 + r + 8 * h;
            float s0 = sc[0][r] * INV_SQRT_HD;
            float s1 = sc[1][r] * INV_SQRT_HD;
            if (key0 + lm > qglob)      s0 = -1e30f;
            if (key0 + 16 + lm > qglob) s1 = -1e30f;

            float rmax = fmaxf(s0, s1);
#pragma unroll
            for (int off = 1; off < 16; off <<= 1)
                rmax = fmaxf(rmax, __shfl_xor(rmax, off, 16));
            const float mnew = fmaxf(m_run[r], rmax);
            const float p0 = __expf(s0 - mnew);
            const float p1 = __expf(s1 - mnew);
            Ps[wave][r + 8 * h][lm]      = (bf16)p0;
            Ps[wave][r + 8 * h][16 + lm] = (bf16)p1;
            float psum = p0 + p1;
#pragma unroll
            for (int off = 1; off < 16; off <<= 1)
                psum += __shfl_xor(psum, off, 16);
            const float alpha = __expf(m_run[r] - mnew);
            l_run[r] = l_run[r] * alpha + psum;
            m_run[r] = mnew;
#pragma unroll
            for (int n = 0; n < 8; ++n) acc[n][r] *= alpha;
        }

        // wave-local LDS visibility for Ps (private per wave)
        asm volatile("s_wait_dscnt 0x0" ::: "memory");

        // P A-frag: row M = lm; i<8 -> key = i+8h ; i>=8 -> key = 16+(i-8)+8h
        v16bf pf = mk16(*(const v8bf*)&Ps[wave][lm][8 * h],
                        *(const v8bf*)&Ps[wave][lm][16 + 8 * h]);

        // O(16x128) += P(16x32) * V(32x128); B element (i,lane): key = i<8 ? i+8h : 16+(i-8)+8h, col d=16n+lm
#pragma unroll
        for (int n = 0; n < 8; ++n) {
            v16bf bf = mk16(*(const v8bf*)&Vst[16 * n + lm][8 * h],
                            *(const v8bf*)&Vst[16 * n + lm][16 + 8 * h]);
            acc[n] = wmma_bf16(pf, bf, acc[n]);
        }
    }
    __syncthreads();

    // normalize and store context as [B*S, H] (col = head*128 + d)
#pragma unroll
    for (int r = 0; r < 8; ++r) {
        const float inv = 1.0f / l_run[r];
        const int srow = qt * 128 + wave * 16 + r + 8 * h;
#pragma unroll
        for (int n = 0; n < 8; ++n) {
            const size_t idx = (size_t)(b * S + srow) * H + head * HD + 16 * n + lm;
            ctx[idx] = (bf16)(acc[n][r] * inv);
        }
    }
}

// ---------------- host side ----------------
extern "C" void kernel_launch(void* const* d_in, const int* in_sizes, int n_in,
                              void* d_out, int out_size, void* d_ws, size_t ws_size,
                              hipStream_t stream) {
    const float* hx = (const float*)d_in[0];
    const float* qw = (const float*)d_in[1];
    const float* kw = (const float*)d_in[2];
    const float* vw = (const float*)d_in[3];
    const float* ow = (const float*)d_in[4];
    // d_in[5] = attention_mask (causal, reproduced analytically)
    // d_in[6] = position_ids (arange, reproduced analytically)

    char* ws = (char*)d_ws;
    constexpr size_t SZ_X = (size_t)M_ROWS * H * sizeof(bf16);   // 16 MiB
    constexpr size_t SZ_W = (size_t)H * H * sizeof(bf16);        // 8 MiB

    bf16* Xb  = (bf16*)(ws);
    bf16* Wqb = (bf16*)(ws + SZ_X);
    bf16* Wkb = (bf16*)(ws + SZ_X + SZ_W);
    bf16* Wvb = (bf16*)(ws + SZ_X + 2 * SZ_W);
    bf16* Wob = (bf16*)(ws + SZ_X + 3 * SZ_W);
    bf16* Qp  = (bf16*)(ws + SZ_X + 4 * SZ_W);
    bf16* Kp  = (bf16*)(ws + 2 * SZ_X + 4 * SZ_W);
    bf16* Vp  = (bf16*)(ws + 3 * SZ_X + 4 * SZ_W);
    bf16* Qr  = (bf16*)(ws + 4 * SZ_X + 4 * SZ_W);
    bf16* Kr  = (bf16*)(ws + 5 * SZ_X + 4 * SZ_W);
    bf16* Vr  = (bf16*)(ws + 6 * SZ_X + 4 * SZ_W);
    bf16* ctx = Qp;  // reuse projection slot after RoPE consumes it

    const int nX = M_ROWS * H;
    const int nW = H * H;
    cvt_f32_bf16<<<nX / 256, 256, 0, stream>>>(hx, Xb, nX);
    cvt_f32_bf16<<<nW / 256, 256, 0, stream>>>(qw, Wqb, nW);
    cvt_f32_bf16<<<nW / 256, 256, 0, stream>>>(kw, Wkb, nW);
    cvt_f32_bf16<<<nW / 256, 256, 0, stream>>>(vw, Wvb, nW);
    cvt_f32_bf16<<<nW / 256, 256, 0, stream>>>(ow, Wob, nW);

    const dim3 gGemm((M_ROWS / 128) * (H / 128));   // 512
    gemm_xwt<false><<<gGemm, 256, 0, stream>>>(Xb, Wqb, (void*)Qp, M_ROWS, H, H);
    gemm_xwt<false><<<gGemm, 256, 0, stream>>>(Xb, Wkb, (void*)Kp, M_ROWS, H, H);
    gemm_xwt<false><<<gGemm, 256, 0, stream>>>(Xb, Wvb, (void*)Vp, M_ROWS, H, H);

    rope_transpose<<<B * NH * S, 128, 0, stream>>>(Qp, Kp, Vp, Qr, Kr, Vr);

    flash_attn<<<B * NH * (S / 128), 256, 0, stream>>>(Qr, Kr, Vr, ctx);

    gemm_xwt<true><<<gGemm, 256, 0, stream>>>(ctx, Wob, d_out, M_ROWS, H, H);
}
